// GIN_pool_10213432229997
// MI455X (gfx1250) — matrix operand order repaired and checked by reading, not compiled, not run on hardware
//
#include <hip/hip_runtime.h>

#define N_NODES  100000
#define N_EDGES  1600000
#define HIDDEN   128
#define N_GRAPHS 512
#define BN_EPS   1e-5f

typedef __attribute__((ext_vector_type(16))) _Float16 v16h;
typedef __attribute__((ext_vector_type(8)))  float    v8f;

// ---------------- aggregation: agg = x + scatter_add(x[src] -> dst) ----------------

__global__ void copy_f4(const float4* __restrict__ src, float4* __restrict__ dst, int n4) {
    int i = blockIdx.x * blockDim.x + threadIdx.x;
    if (i < n4) dst[i] = src[i];
}

__global__ void edge_scatter(const float* __restrict__ x, const int* __restrict__ ei,
                             float* __restrict__ agg) {
    long long t = (long long)blockIdx.x * blockDim.x + threadIdx.x;
    long long edge = t >> 5;           // 32 lanes per edge, 4 floats per lane
    int q = (int)(t & 31);
    if (edge >= N_EDGES) return;
    int s = ei[edge];
    int d = ei[N_EDGES + edge];
    const float4 v = *(const float4*)(x + (size_t)s * HIDDEN + q * 4);
    float* o = agg + (size_t)d * HIDDEN + q * 4;
    atomicAdd(o + 0, v.x);
    atomicAdd(o + 1, v.y);
    atomicAdd(o + 2, v.z);
    atomicAdd(o + 3, v.w);
}

// ---------------- fused GEMM + epilogue (WMMA f16 -> f32 accum) ----------------
// in:[N,128] f32, W:[128,128] f32 (k-major). Each wave: 16 rows x 128 cols.
// use_bn=1: out = bn(relu(in@W + bias)) ; use_bn=0: out = relu(in@W + bias) (+resid)

__global__ __launch_bounds__(256)
void gin_gemm(const float* __restrict__ in, const float* __restrict__ W,
              const float* __restrict__ bias,
              const float* __restrict__ gamma, const float* __restrict__ beta,
              const float* __restrict__ mean,  const float* __restrict__ var,
              const float* __restrict__ resid,
              float* __restrict__ out, int use_bn) {
    __shared__ _Float16 sW[HIDDEN * HIDDEN];   // transposed: sW[n*128 + k], 32 KB
    for (int i = threadIdx.x; i < HIDDEN * HIDDEN; i += 256) {
        int n = i >> 7, k = i & 127;
        sW[i] = (_Float16)W[k * HIDDEN + n];
    }
    __syncthreads();

    const int lane = threadIdx.x & 31;
    const int wv   = threadIdx.x >> 5;
    const int m    = lane & 15;        // row within tile (A) / col within tile (B,C)
    const int h    = lane >> 4;        // half-wave select
    const int row0 = blockIdx.x * 128 + wv * 16;
    const int arow = row0 + m;

    v8f acc[8];
    #pragma unroll
    for (int nt = 0; nt < 8; ++nt)
        #pragma unroll
        for (int r = 0; r < 8; ++r) acc[nt][r] = 0.0f;

    #pragma unroll
    for (int kt = 0; kt < 4; ++kt) {
        const int kbase = kt * 32;
        v16h a;
        if (arow < N_NODES) {
            // A 16x32 f16: lane holds K = {16h..16h+15}
            const float* p = in + (size_t)arow * HIDDEN + kbase + 16 * h;
            #pragma unroll
            for (int j = 0; j < 8; ++j) a[j]     = (_Float16)p[j];
            #pragma unroll
            for (int j = 0; j < 8; ++j) a[8 + j] = (_Float16)p[8 + j];
        } else {
            #pragma unroll
            for (int j = 0; j < 16; ++j) a[j] = (_Float16)0.0f;
        }
        #pragma unroll
        for (int nt = 0; nt < 8; ++nt) {
            // B 32x16 f16: lane = column n, holds 16 contiguous K (32 B from LDS)
            const _Float16* bp = &sW[(size_t)(nt * 16 + m) * HIDDEN + kbase + 16 * h];
            v16h b;
            #pragma unroll
            for (int j = 0; j < 16; ++j) b[j] = bp[j];
            acc[nt] = __builtin_amdgcn_wmma_f32_16x16x32_f16(
                false, a, false, b, (short)0, acc[nt], false, false);
        }
    }

    // fused epilogue
    #pragma unroll
    for (int nt = 0; nt < 8; ++nt) {
        const int n = nt * 16 + m;
        const float bv = bias[n];
        float scale = 1.0f, shift = 0.0f;
        if (use_bn) {
            const float sc = gamma[n] * rsqrtf(var[n] + BN_EPS);
            scale = sc;
            shift = beta[n] - mean[n] * sc;
        }
        #pragma unroll
        for (int r = 0; r < 8; ++r) {
            const int rr = row0 + 8 * h + r;   // C layout: M = r + 8*h
            if (rr < N_NODES) {
                float v = acc[nt][r] + bv;
                v = fmaxf(v, 0.0f);
                if (use_bn) v = v * scale + shift;
                if (resid)  v += resid[(size_t)rr * HIDDEN + n];
                out[(size_t)rr * HIDDEN + n] = v;
            }
        }
    }
}

// ---------------- pooling + final linear ----------------

__global__ void zero_f(float* __restrict__ p, int n) {
    int i = blockIdx.x * blockDim.x + threadIdx.x;
    if (i < n) p[i] = 0.0f;
}

__global__ void pool_scatter(const float* __restrict__ x, const int* __restrict__ batch,
                             float* __restrict__ sums, float* __restrict__ counts) {
    long long t = (long long)blockIdx.x * blockDim.x + threadIdx.x;
    long long node = t >> 5;
    int q = (int)(t & 31);
    if (node >= N_NODES) return;
    int g = batch[node];
    const float4 v = *(const float4*)(x + (size_t)node * HIDDEN + q * 4);
    float* o = sums + (size_t)g * HIDDEN + q * 4;
    atomicAdd(o + 0, v.x);
    atomicAdd(o + 1, v.y);
    atomicAdd(o + 2, v.z);
    atomicAdd(o + 3, v.w);
    if (q == 0) atomicAdd(&counts[g], 1.0f);
}

__global__ void final_lin(const float* __restrict__ sums, const float* __restrict__ counts,
                          const float* __restrict__ lin_w, const float* __restrict__ lin_b,
                          float* __restrict__ out) {
    int g = blockIdx.x * (blockDim.x >> 5) + (threadIdx.x >> 5);
    int lane = threadIdx.x & 31;
    if (g >= N_GRAPHS) return;
    float c = fmaxf(counts[g], 1.0f);
    float a = 0.0f;
    for (int j = lane; j < HIDDEN; j += 32)
        a += sums[(size_t)g * HIDDEN + j] * lin_w[j];
    #pragma unroll
    for (int off = 16; off; off >>= 1) a += __shfl_down(a, off);
    if (lane == 0) out[g] = a / c + lin_b[0];
}

// ---------------- host orchestration ----------------

extern "C" void kernel_launch(void* const* d_in, const int* in_sizes, int n_in,
                              void* d_out, int out_size, void* d_ws, size_t ws_size,
                              hipStream_t stream) {
    (void)in_sizes; (void)n_in; (void)out_size; (void)ws_size;
    const float* x     = (const float*)d_in[0];
    const int*   ei    = (const int*)d_in[1];
    const int*   batch = (const int*)d_in[2];
    const float* W1s   = (const float*)d_in[3];
    const float* b1s   = (const float*)d_in[4];
    const float* W2s   = (const float*)d_in[5];
    const float* b2s   = (const float*)d_in[6];
    const float* gam   = (const float*)d_in[7];
    const float* bet   = (const float*)d_in[8];
    const float* mu    = (const float*)d_in[9];
    const float* var   = (const float*)d_in[10];
    const float* lw    = (const float*)d_in[11];
    const float* lb    = (const float*)d_in[12];
    float* out = (float*)d_out;

    const size_t FEAT = (size_t)N_NODES * HIDDEN;
    float* B0 = (float*)d_ws;
    float* B1 = B0 + FEAT;
    float* B2 = B1 + FEAT;
    float* Hb = B2 + FEAT;
    float* Bbuf[3] = {B0, B1, B2};

    const float* cur = x;  int curI = -1;   // -1 == input x (read-only)
    const float* x0  = x;  int x0I  = -1;

    const dim3 blk(256);
    const int n4 = (int)(FEAT / 4);
    const int copyBlocks = (n4 + 255) / 256;
    const long long scatThreads = (long long)N_EDGES * 32;
    const int scatBlocks = (int)((scatThreads + 255) / 256);
    const int gemmBlocks = (N_NODES + 127) / 128;

    for (int l = 0; l < 6; ++l) {
        int t = 0;
        while (t == curI || t == x0I) ++t;   // pick buffer distinct from cur and x0
        float* T = Bbuf[t];

        // agg = cur + scatter_add(cur[src] -> dst)
        copy_f4<<<copyBlocks, blk, 0, stream>>>((const float4*)cur, (float4*)T, n4);
        edge_scatter<<<scatBlocks, blk, 0, stream>>>(cur, ei, T);

        // h = bn(relu(agg @ W1 + b1))
        gin_gemm<<<gemmBlocks, blk, 0, stream>>>(
            T, W1s + (size_t)l * HIDDEN * HIDDEN, b1s + l * HIDDEN,
            gam + l * HIDDEN, bet + l * HIDDEN, mu + l * HIDDEN, var + l * HIDDEN,
            nullptr, Hb, 1);

        // xnew = relu(h @ W2 + b2) (+ x0 residual on odd layers); overwrite agg buf
        const float* res = (l & 1) ? x0 : nullptr;
        gin_gemm<<<gemmBlocks, blk, 0, stream>>>(
            Hb, W2s + (size_t)l * HIDDEN * HIDDEN, b2s + l * HIDDEN,
            nullptr, nullptr, nullptr, nullptr, res, T, 0);

        cur = T; curI = t;
        if (l & 1) { x0 = T; x0I = t; }
    }

    // global mean pool + final linear (reuse Hb for sums/counts; re-zero every call)
    float* sums   = Hb;
    float* counts = Hb + (size_t)N_GRAPHS * HIDDEN;
    const int zn = N_GRAPHS * HIDDEN + N_GRAPHS;
    zero_f<<<(zn + 255) / 256, blk, 0, stream>>>(sums, zn);
    const long long poolThreads = (long long)N_NODES * 32;
    pool_scatter<<<(int)((poolThreads + 255) / 256), blk, 0, stream>>>(cur, batch, sums, counts);
    final_lin<<<(N_GRAPHS + 7) / 8, blk, 0, stream>>>(sums, counts, lw, lb, out);
}
